// LightGCN_57423712748257
// MI455X (gfx1250) — matrix operand order repaired and checked by reading, not compiled, not run on hardware
//
#include <hip/hip_runtime.h>

#define N_USER 200000
#define N_ITEM 100000
#define N_NODE (N_USER + N_ITEM)
#define DIM 64
#define NNZ_E 2000000
#define N_LAYERS 3
#define BATCH_N 8192

typedef __attribute__((ext_vector_type(16))) __bf16 v16bf;
typedef __attribute__((ext_vector_type(8)))  float  v8f;

// Guaranteed-hardware f32 atomic add (no return -> STOREcnt path, resolves in L2).
__device__ __forceinline__ void atomic_add_f32_hw(float* addr, float val) {
  asm volatile("global_atomic_add_f32 %0, %1, off"
               :
               : "v"(addr), "v"(val)
               : "memory");
}

// cur = concat(emb_u, emb_i); acc = cur   (float4-vectorized)
__global__ void lgcn_init(const float* __restrict__ eu, const float* __restrict__ ei,
                          float4* __restrict__ cur, float4* __restrict__ acc) {
  long i = (long)blockIdx.x * blockDim.x + threadIdx.x;   // float4 index
  const long total  = (long)N_NODE * (DIM / 4);
  const long ucount = (long)N_USER * (DIM / 4);
  if (i >= total) return;
  float4 v = (i < ucount) ? ((const float4*)eu)[i]
                          : ((const float4*)ei)[i - ucount];
  cur[i] = v;
  acc[i] = v;
}

__global__ void lgcn_zero(float4* __restrict__ p, long n4) {
  long i = (long)blockIdx.x * blockDim.x + threadIdx.x;
  if (i >= n4) return;
  p[i] = make_float4(0.f, 0.f, 0.f, 0.f);
}

// COO SpMM: one wave (32 lanes) per edge, float2 per lane -> coalesced 256B
// gather of x[col] (L2-resident: table is 76.8MB < 192MB L2) and coalesced
// hardware global_atomic_add_f32 scatter into y[row].
__global__ void lgcn_spmm(const int* __restrict__ rows, const int* __restrict__ cols,
                          const float* __restrict__ vals,
                          const float* __restrict__ x, float* __restrict__ y) {
  long tid  = (long)blockIdx.x * blockDim.x + threadIdx.x;
  long edge = tid >> 5;
  int  lane = (int)(tid & 31);
  if (edge >= NNZ_E) return;
  int   r = rows[edge];
  int   c = cols[edge];
  float v = vals[edge];
  float2 xv = ((const float2*)(x + (long)c * DIM))[lane];
  float* yr = y + (long)r * DIM + lane * 2;
  atomic_add_f32_hw(yr,     v * xv.x);
  atomic_add_f32_hw(yr + 1, v * xv.y);
}

// acc += nxt   (streaming, float4)
__global__ void lgcn_accadd(float4* __restrict__ acc, const float4* __restrict__ xn, long n4) {
  long i = (long)blockIdx.x * blockDim.x + threadIdx.x;
  if (i >= n4) return;
  float4 a = acc[i], b = xn[i];
  a.x += b.x; a.y += b.y; a.z += b.z; a.w += b.w;
  acc[i] = a;
}

// Batched dot products via WMMA: each wave handles 16 batch pairs.
// A = 16 gathered user rows (16x64), B = 16 gathered item rows as columns
// (64x16); two v_wmma_f32_16x16x32_bf16 (K=0..31, K=32..63) accumulate the
// 16x16 cross-product matrix; we extract the diagonal. The 1/(L+1)=1/4 mean
// on each side is folded into a final *1/16.
__global__ void lgcn_dot_wmma(const float* __restrict__ acc,
                              const int* __restrict__ idx_u,
                              const int* __restrict__ idx_i,
                              float* __restrict__ out) {
  int tid  = blockIdx.x * blockDim.x + threadIdx.x;
  int g    = tid >> 5;            // group of 16 batch elements per wave
  int lane = tid & 31;
  if (g >= BATCH_N / 16) return;  // uniform; grid is exact so never taken

  int slot = lane & 15;
  int hi   = lane >> 4;           // 0: lanes 0-15, 1: lanes 16-31
  const float* arow = acc + (long)idx_u[g * 16 + slot] * DIM;            // A row M=slot
  const float* brow = acc + (long)(idx_i[g * 16 + slot] + N_USER) * DIM; // B col N=slot

  v8f C = {};
  #pragma unroll
  for (int chunk = 0; chunk < 2; ++chunk) {   // K blocks [0,32) and [32,64)
    v16bf A, B;
    #pragma unroll
    for (int e = 0; e < 16; ++e) {
      // A 16x32 bf16 layout: lane<16 holds K {0..7,16..23}, lane>=16 {8..15,24..31}
      int ka = (e < 8) ? (hi * 8 + e) : (8 + hi * 8 + e);
      // B 32x16 bf16 layout: same K order as A: lane<16 holds K {0..7,16..23}, lane>=16 {8..15,24..31}
      int kb = (e < 8) ? (hi * 8 + e) : (8 + hi * 8 + e);
      A[e] = (__bf16)arow[chunk * 32 + ka];
      B[e] = (__bf16)brow[chunk * 32 + kb];
    }
    C = __builtin_amdgcn_wmma_f32_16x16x32_bf16(
        /*neg_a=*/false, A, /*neg_b=*/false, B,
        /*c_mod=*/(short)0, C, /*reuse_a=*/false, /*reuse_b=*/false);
  }

  // D diag: lanes 0-15 hold M=0..7 in VGPR0..7 (N=lane); lanes 16-31 hold M=8..15.
  int n = lane & 15;
  if (hi == (n >> 3)) {
    out[g * 16 + n] = C[n & 7] * 0.0625f;   // (1/4)*(1/4) layer means
  }
}

extern "C" void kernel_launch(void* const* d_in, const int* in_sizes, int n_in,
                              void* d_out, int out_size, void* d_ws, size_t ws_size,
                              hipStream_t stream) {
  const float* emb_u = (const float*)d_in[0];
  const float* emb_i = (const float*)d_in[1];
  const int*   rows  = (const int*)  d_in[2];
  const int*   cols  = (const int*)  d_in[3];
  const float* vals  = (const float*)d_in[4];
  const int*   idx_u = (const int*)  d_in[5];
  const int*   idx_i = (const int*)  d_in[6];
  float*       out   = (float*)d_out;

  const size_t nfloats = (size_t)N_NODE * DIM;     // 19.2M floats = 76.8 MB
  float* acc  = (float*)d_ws;                      // ws layout: acc | buf0 | buf1
  float* buf0 = acc  + nfloats;
  float* buf1 = buf0 + nfloats;

  const long n4 = (long)N_NODE * (DIM / 4);
  const int  tb = 256;
  const long nblk = (n4 + tb - 1) / tb;

  lgcn_init<<<(unsigned)nblk, tb, 0, stream>>>(emb_u, emb_i, (float4*)buf0, (float4*)acc);

  float* cur = buf0;
  float* nxt = buf1;
  for (int l = 0; l < N_LAYERS; ++l) {
    lgcn_zero<<<(unsigned)nblk, tb, 0, stream>>>((float4*)nxt, n4);
    const long spmm_threads = (long)NNZ_E * 32;
    lgcn_spmm<<<(unsigned)((spmm_threads + tb - 1) / tb), tb, 0, stream>>>(rows, cols, vals, cur, nxt);
    lgcn_accadd<<<(unsigned)nblk, tb, 0, stream>>>((float4*)acc, (const float4*)nxt, n4);
    float* t = cur; cur = nxt; nxt = t;
  }

  const int dot_threads = (BATCH_N / 16) * 32;     // 512 waves, EXEC fully populated
  lgcn_dot_wmma<<<dot_threads / tb, tb, 0, stream>>>(acc, idx_u, idx_i, out);
}
